// CoxLoss_1443109012355
// MI455X (gfx1250) — matrix operand order repaired and checked by reading, not compiled, not run on hardware
//
#include <hip/hip_runtime.h>
#include <hip/hip_bf16.h>

typedef __attribute__((ext_vector_type(16))) _Float16 v16h;
typedef __attribute__((ext_vector_type(8)))  float    v8f;
typedef __attribute__((ext_vector_type(4)))  float    v4f;

#define COX_EPS 1e-7f

// ---------------------------------------------------------------------------
// Kernel 0: zero the two global accumulators (sum_partial, sum_events)
// ---------------------------------------------------------------------------
__global__ void cox_init_kernel(float* acc) {
    acc[0] = 0.0f;
    acc[1] = 0.0f;
}

// ---------------------------------------------------------------------------
// Kernel 1: exp_h[j] = (f16) exp(r[j]); block-reduce event indicators into
// acc[1] (sum of events).
// ---------------------------------------------------------------------------
__global__ void cox_prep_kernel(const float* __restrict__ r,
                                const float* __restrict__ ev,
                                _Float16* __restrict__ exp_h,
                                float* __restrict__ acc, int N) {
    int i = blockIdx.x * blockDim.x + threadIdx.x;
    float e_val = 0.0f;
    if (i < N) {
        exp_h[i] = (_Float16)__expf(r[i]);
        e_val = ev[i];
    }
    __shared__ float sm[256];
    sm[threadIdx.x] = e_val;
    __syncthreads();
    for (int s = 128; s > 0; s >>= 1) {
        if ((int)threadIdx.x < s) sm[threadIdx.x] += sm[threadIdx.x + s];
        __syncthreads();
    }
    if (threadIdx.x == 0) atomicAdd(&acc[1], sm[0]);
}

// ---------------------------------------------------------------------------
// Kernel 2: WMMA main pass. One wave32 per TWO 16-row tiles (32 rows).
// A[m,k] = (t[row m] >= t[col k]) as f16 {0,1};  B[k,n] = exp_h[col k]
// (all columns identical).  D = A*B + C accumulates
//   S[i] = sum_j mask[i,j] * exp(r[j])
// in f32 across N/32 chunks of K=32.  Each loaded (t, exp) chunk feeds two
// WMMAs (two row tiles) -> half the VMEM per matrix op, two independent
// WMMA dep chains that co-execute with the VALU mask build.
//
// VGPR layouts (CDNA5 ISA 7.12.2, wave32):
//   A 16x32 f16 : lane = 16*half + m ; element e: K = e + 8*half + (e>=8 ? 8:0)
//   B 32x16 f16 : element e: K = e + 8*half + (e>=8 ? 8:0)  (column = lane&15, irrelevant
//                 here since all columns equal) -> two 16B loads
//   C/D 16x16 f32: VGPR q -> M = q + 8*half, column N = lane&15
// ---------------------------------------------------------------------------
__global__ void cox_wmma_kernel(const float* __restrict__ t,
                                const _Float16* __restrict__ exp_h,
                                const float* __restrict__ r,
                                const float* __restrict__ ev,
                                float* __restrict__ acc, int N) {
    const int gtid = blockIdx.x * blockDim.x + threadIdx.x;
    const int wave = gtid >> 5;          // one wave per 32-row super-tile
    const int lane = threadIdx.x & 31;
    const int half = lane >> 4;          // 0: lanes 0-15, 1: lanes 16-31
    const int m    = lane & 15;

    const int row0 = wave * 32;          // first row of this super-tile
    const float ti0 = t[row0 + m];       // row time, tile 0
    const float ti1 = t[row0 + 16 + m];  // row time, tile 1

    const _Float16 h1 = (_Float16)1.0f;
    const _Float16 h0 = (_Float16)0.0f;

    v8f s_acc0 = {};
    v8f s_acc1 = {};

    const int chunks = N >> 5;           // N / 32
    for (int c = 0; c < chunks; ++c) {
        const int base = c << 5;

        // --- shared column data for both tiles -----------------------------
        // A elements 0..7  -> t[base + 8*half + 0..7]
        // A elements 8..15 -> t[base + 16 + 8*half + 0..7]
        const v4f* tp0 = (const v4f*)(t + base + 8 * half);
        const v4f* tp1 = (const v4f*)(t + base + 16 + 8 * half);
        v4f t0 = tp0[0], t1 = tp0[1];
        v4f t2 = tp1[0], t3 = tp1[1];

        // B operand: broadcast exp vector, element e -> K = e + 8*half + (e>=8 ? 8:0)
        typedef __attribute__((ext_vector_type(8))) _Float16 v8h_;
        v8h_ blo = *(const v8h_*)(exp_h + base + 8 * half);
        v8h_ bhi = *(const v8h_*)(exp_h + base + 16 + 8 * half);
        v16h b = __builtin_shufflevector(blo, bhi, 0,1,2,3,4,5,6,7,8,9,10,11,12,13,14,15);

        // --- tile 0 mask + WMMA -------------------------------------------
        v16h a0;
#pragma unroll
        for (int e = 0; e < 4; ++e) a0[e]      = (ti0 >= t0[e]) ? h1 : h0;
#pragma unroll
        for (int e = 0; e < 4; ++e) a0[e + 4]  = (ti0 >= t1[e]) ? h1 : h0;
#pragma unroll
        for (int e = 0; e < 4; ++e) a0[e + 8]  = (ti0 >= t2[e]) ? h1 : h0;
#pragma unroll
        for (int e = 0; e < 4; ++e) a0[e + 12] = (ti0 >= t3[e]) ? h1 : h0;

        s_acc0 = __builtin_amdgcn_wmma_f32_16x16x32_f16(
            false, a0, false, b, (short)0, s_acc0, false, false);

        // --- tile 1 mask + WMMA (independent chain) -----------------------
        v16h a1;
#pragma unroll
        for (int e = 0; e < 4; ++e) a1[e]      = (ti1 >= t0[e]) ? h1 : h0;
#pragma unroll
        for (int e = 0; e < 4; ++e) a1[e + 4]  = (ti1 >= t1[e]) ? h1 : h0;
#pragma unroll
        for (int e = 0; e < 4; ++e) a1[e + 8]  = (ti1 >= t2[e]) ? h1 : h0;
#pragma unroll
        for (int e = 0; e < 4; ++e) a1[e + 12] = (ti1 >= t3[e]) ? h1 : h0;

        s_acc1 = __builtin_amdgcn_wmma_f32_16x16x32_f16(
            false, a1, false, b, (short)0, s_acc1, false, false);
    }

    // --- extract row sums from column 0 (lanes 0 and 16), finish per-row ---
    if ((lane & 15) == 0) {
        float local = 0.0f;
        const int rb0 = row0 + 8 * half;        // tile 0: VGPR q -> M = q+8*half
        const int rb1 = row0 + 16 + 8 * half;   // tile 1
#pragma unroll
        for (int q = 0; q < 8; ++q) {
            const float S0 = s_acc0[q];
            const int row_a = rb0 + q;
            local += ev[row_a] * (r[row_a] - __logf(S0 + COX_EPS));
        }
#pragma unroll
        for (int q = 0; q < 8; ++q) {
            const float S1 = s_acc1[q];
            const int row_b = rb1 + q;
            local += ev[row_b] * (r[row_b] - __logf(S1 + COX_EPS));
        }
        atomicAdd(&acc[0], local);
    }
}

// ---------------------------------------------------------------------------
// Kernel 3: loss = -sum_partial / (sum_events + EPS)
// ---------------------------------------------------------------------------
__global__ void cox_final_kernel(const float* __restrict__ acc,
                                 float* __restrict__ out) {
    out[0] = -acc[0] / (acc[1] + COX_EPS);
}

// ---------------------------------------------------------------------------
extern "C" void kernel_launch(void* const* d_in, const int* in_sizes, int n_in,
                              void* d_out, int out_size, void* d_ws, size_t ws_size,
                              hipStream_t stream) {
    const float* r  = (const float*)d_in[0];  // risk_scores
    const float* t  = (const float*)d_in[1];  // survival_time
    const float* ev = (const float*)d_in[2];  // event_indicator
    float* out = (float*)d_out;
    const int N = in_sizes[0];                // 16384 (multiple of 32)

    // Workspace layout: [0, N*2) f16 exp table (32B-aligned chunks),
    // then two f32 accumulators {sum_partial, sum_events}.
    _Float16* exp_h = (_Float16*)d_ws;
    float* acc = (float*)((char*)d_ws + (size_t)N * sizeof(_Float16));

    cox_init_kernel<<<1, 1, 0, stream>>>(acc);

    cox_prep_kernel<<<(N + 255) / 256, 256, 0, stream>>>(r, ev, exp_h, acc, N);

    // One wave32 per 32-row super-tile (two 16x16 WMMA row tiles):
    // N/32 waves, 8 waves (256 threads) per block.
    const int waves = N / 32;
    const int threads = 256;
    const int blocks = (waves * 32) / threads;
    cox_wmma_kernel<<<blocks, threads, 0, stream>>>(t, exp_h, r, ev, acc, N);

    cox_final_kernel<<<1, 1, 0, stream>>>(acc, out);
}
